// GCNTox21NNConv_60120952209752
// MI455X (gfx1250) — matrix operand order repaired and checked by reading, not compiled, not run on hardware
//
#include <hip/hip_runtime.h>
#include <hip/hip_bf16.h>
#include <cmath>

typedef __attribute__((ext_vector_type(16))) _Float16 v16h;
typedef __attribute__((ext_vector_type(8)))  _Float16 v8h;
typedef __attribute__((ext_vector_type(8)))  float    v8f;

static constexpr int N = 16384;
static constexpr int E = 65536;
static constexpr int G = 512;
static constexpr float EPS = 1e-5f;

// GEMM geometry: K_total = 128*64 (kron of ew[128] x h[64]) + 64 (b2 bias rows)
static constexpr int KSTEPS   = 258;   // 256 weight K-steps + 2 bias steps
static constexpr int CH       = 16;    // K-steps per LDS chunk
static constexpr int NCH_W    = 16;    // full weight chunks
static constexpr int WAVES    = 8;     // waves per workgroup
static constexpr int MT_W     = 3;     // 16-edge M-tiles per wave (48 edges/wave)
static constexpr int NT_TOT   = E / 16;  // 4096 M-tiles

__device__ __forceinline__ int imax(int a, int b) { return a > b ? a : b; }

// ---------------- small dense kernels ----------------

// ee[e,j] = edge_attr[e,0:3] . ee_w[:,j] + ee_b[j]       (E x 32)
__global__ void k_edge_embed(const float* __restrict__ ea, const float* __restrict__ w,
                             const float* __restrict__ b, float* __restrict__ ee) {
  int idx = blockIdx.x * blockDim.x + threadIdx.x;
  if (idx >= E * 32) return;
  int j = idx & 31, e = idx >> 5;
  float acc = b[j];
#pragma unroll
  for (int k = 0; k < 3; ++k) acc += ea[e * 3 + k] * w[k * 32 + j];
  ee[idx] = acc;
}

// h[n,j] = x[n,0:9] . ne_w[:,j] + ne_b[j]                (N x 64), also f16 copy
__global__ void k_node_embed(const float* __restrict__ x, const float* __restrict__ w,
                             const float* __restrict__ b, float* __restrict__ h,
                             _Float16* __restrict__ h16) {
  int idx = blockIdx.x * blockDim.x + threadIdx.x;
  if (idx >= N * 64) return;
  int j = idx & 63, n = idx >> 6;
  float acc = b[j];
#pragma unroll
  for (int k = 0; k < 9; ++k) acc += x[n * 9 + k] * w[k * 64 + j];
  h[idx] = acc;
  h16[idx] = (_Float16)acc;
}

__global__ void k_deg(const int* __restrict__ dst, int* __restrict__ deg) {
  int idx = blockIdx.x * blockDim.x + threadIdx.x;
  if (idx < E) atomicAdd(&deg[dst[idx]], 1);
}
__global__ void k_cnt(const int* __restrict__ batch, int* __restrict__ cnt) {
  int idx = blockIdx.x * blockDim.x + threadIdx.x;
  if (idx < N) atomicAdd(&cnt[batch[idx]], 1);
}

// ew[e,j] = relu(ee[e,:] @ w1[:,j] + b1[j])  -> f16      (E x 128)
__global__ void k_edge_mlp(const float* __restrict__ ee, const float* __restrict__ w1,
                           const float* __restrict__ b1, _Float16* __restrict__ ew) {
  int idx = blockIdx.x * blockDim.x + threadIdx.x;
  if (idx >= E * 128) return;
  int j = idx & 127, e = idx >> 7;
  const float* r = ee + (size_t)e * 32;
  float acc = b1[j];
#pragma unroll
  for (int k = 0; k < 32; ++k) acc += r[k] * w1[k * 128 + j];
  ew[idx] = (_Float16)fmaxf(acc, 0.f);
}

// Pre-pack extended weight matrix (kron-K x dout) into WMMA B-fragment order:
// Bp[t][nt][lane][tt] = B[kk = t*32 + (tt>>3)*16 + ((lane>=16)?8:0) + (tt&7)][col = nt*16 + (lane&15)]
// where B[kk,c] = w2[kk>>6, (kk&63)*dout + c] for kk<8192, else b2[(kk-8192)*dout + c]
__global__ void k_repack(const float* __restrict__ w2, const float* __restrict__ b2,
                         _Float16* __restrict__ Bp, int ntiles, int dout) {
  int idx = blockIdx.x * blockDim.x + threadIdx.x;
  int total = KSTEPS * ntiles * 512;
  if (idx >= total) return;
  int tt   = idx & 15;
  int lane = (idx >> 4) & 31;
  int nt   = (idx >> 9) % ntiles;
  int t    = idx / (512 * ntiles);
  int col  = nt * 16 + (lane & 15);
  int kap  = ((tt >> 3) << 4) + ((lane >> 4) << 3) + (tt & 7);
  int kk   = t * 32 + kap;
  float v;
  if (kk < 8192) {
    int k = kk >> 6, i = kk & 63;
    v = w2[(size_t)k * (64 * dout) + i * dout + col];
  } else {
    int i = kk - 8192;
    v = b2[i * dout + col];
  }
  Bp[idx] = (_Float16)v;
}

// ---------------- fused kron-GEMM message kernel ----------------
// msg[e, :] = (ew[e] (x) h16[src[e]]) @ Wr + h16[src[e]] @ B2, scatter-add to aggr[dst[e]]
template <int NTILES>
__global__ __launch_bounds__(256) void k_msg_gemm(
    const _Float16* __restrict__ ew,   // [E,128]
    const _Float16* __restrict__ h16,  // [N,64]
    const _Float16* __restrict__ Bp,   // [KSTEPS][NTILES][512]
    const int* __restrict__ src, const int* __restrict__ dst,
    float* __restrict__ aggr)          // [N, NTILES*16]
{
  constexpr int DOUT = NTILES * 16;
  __shared__ _Float16 lds[CH * NTILES * 512];

  const int tid   = threadIdx.x;
  const int wave  = tid >> 5;
  const int lane  = tid & 31;
  const int m     = lane & 15;
  const int hi    = lane >> 4;       // lane half: selects K sub-bank
  const int kbase = hi * 8;          // A-layout K offset per ISA table

  const int tile0 = blockIdx.x * (WAVES * MT_W) + wave * MT_W;

  v8f  acc[MT_W][NTILES];
  v16h hf[MT_W][2];
  int  e[MT_W];
  bool act[MT_W];                    // wave-uniform activity per m-tile

#pragma unroll
  for (int mt = 0; mt < MT_W; ++mt) {
    const int tl = tile0 + mt;
    act[mt] = (tl < NT_TOT);
    const int tc = act[mt] ? tl : 0;
    e[mt] = tc * 16 + m;
    int s = src[e[mt]];
    const _Float16* hpp = h16 + (size_t)s * 64;
#pragma unroll
    for (int p = 0; p < 2; ++p) {       // i0 = 0 / 32 halves of the h row
      v8h lo = *(const v8h*)(hpp + p * 32 + kbase);
      v8h hh = *(const v8h*)(hpp + p * 32 + kbase + 16);
      v16h f;
#pragma unroll
      for (int t = 0; t < 8; ++t) { f[t] = lo[t]; f[8 + t] = hh[t]; }
      hf[mt][p] = f;
    }
#pragma unroll
    for (int nt = 0; nt < NTILES; ++nt) acc[mt][nt] = (v8f){0, 0, 0, 0, 0, 0, 0, 0};
  }

  // ---- 16 full weight chunks; all fragment indices compile-time ----
  for (int c = 0; c < NCH_W; ++c) {
    const int t0 = c * CH;
    { // cooperative stage of B chunk into LDS
      const uint4* gsrc = (const uint4*)(Bp + (size_t)t0 * NTILES * 512);
      uint4* l = (uint4*)lds;
#pragma unroll
      for (int i = 0; i < CH * NTILES * 64 / 256; ++i) l[tid + i * 256] = gsrc[tid + i * 256];
    }
    __syncthreads();

    v8h ewc[MT_W];
#pragma unroll
    for (int mt = 0; mt < MT_W; ++mt)
      ewc[mt] = *(const v8h*)(ew + (size_t)e[mt] * 128 + (t0 >> 1));

#pragma unroll
    for (int half = 0; half < 8; ++half) {   // each half = 2 K-steps sharing ew scalar
#pragma unroll
      for (int p = 0; p < 2; ++p) {          // p=0: i-range [0,32); p=1: [32,64)
        const int tl = half * 2 + p;
        v16h bfr[NTILES];
#pragma unroll
        for (int nt = 0; nt < NTILES; ++nt)
          bfr[nt] = *(const v16h*)(&lds[(tl * NTILES + nt) * 512 + lane * 16]);
#pragma unroll
        for (int mt = 0; mt < MT_W; ++mt) {
          const _Float16 s = ewc[mt][half];  // compile-time extract
          v16h sv;
#pragma unroll
          for (int q = 0; q < 16; ++q) sv[q] = s;
          const v16h a = hf[mt][p] * sv;     // v_pk_mul_f16 x8 (op_sel broadcast)
#pragma unroll
          for (int nt = 0; nt < NTILES; ++nt)
            acc[mt][nt] = __builtin_amdgcn_wmma_f32_16x16x32_f16(
                false, a, false, bfr[nt], (short)0, acc[mt][nt], false, false);
        }
      }
    }
    __syncthreads();
  }

  // ---- bias tail: 2 K-steps with A = h fragment (scale 1) ----
  {
    const int t0 = NCH_W * CH;  // 256
    {
      const uint4* gsrc = (const uint4*)(Bp + (size_t)t0 * NTILES * 512);
      uint4* l = (uint4*)lds;
      const int n16 = 2 * NTILES * 64;
      for (int i = tid; i < n16; i += 256) l[i] = gsrc[i];
    }
    __syncthreads();
#pragma unroll
    for (int p = 0; p < 2; ++p) {
      v16h bfr[NTILES];
#pragma unroll
      for (int nt = 0; nt < NTILES; ++nt)
        bfr[nt] = *(const v16h*)(&lds[(p * NTILES + nt) * 512 + lane * 16]);
#pragma unroll
      for (int mt = 0; mt < MT_W; ++mt)
#pragma unroll
        for (int nt = 0; nt < NTILES; ++nt)
          acc[mt][nt] = __builtin_amdgcn_wmma_f32_16x16x32_f16(
              false, hf[mt][p], false, bfr[nt], (short)0, acc[mt][nt], false, false);
    }
  }

  // C/D layout: lane holds rows (hi*8 + v), col = lane&15 ; scatter-add by dst
#pragma unroll
  for (int mt = 0; mt < MT_W; ++mt) {
    if (!act[mt]) continue;            // wave-uniform guard
    const int ebase = (tile0 + mt) * 16;
#pragma unroll
    for (int v = 0; v < 8; ++v) {
      int d = dst[ebase + hi * 8 + v];
      float* out = aggr + (size_t)d * DOUT + m;
#pragma unroll
      for (int nt = 0; nt < NTILES; ++nt)
        unsafeAtomicAdd(out + nt * 16, acc[mt][nt][v]);
    }
  }
}

// ---------------- node update + batchnorm ----------------

// one block per output channel o: hp[n,o] = h[n,:]@root[:,o] + aggr[n,o]/deg + cb[o]
// then block-reduce mean/var -> ab[o] (scale), ab[64+o] (shift)
__global__ __launch_bounds__(256) void k_node_update(
    const float* __restrict__ h, const float* __restrict__ aggr,
    const int* __restrict__ deg, const float* __restrict__ root,
    const float* __restrict__ cb, const float* __restrict__ bn_g,
    const float* __restrict__ bn_b, float* __restrict__ hp,
    float* __restrict__ ab, int dout) {
  const int o = blockIdx.x;
  __shared__ float s1[256], s2[256];
  float ls = 0.f, lq = 0.f;
  for (int n = threadIdx.x; n < N; n += 256) {
    float acc = cb[o];
    const float* r = h + (size_t)n * 64;
#pragma unroll 8
    for (int i = 0; i < 64; ++i) acc += r[i] * root[i * dout + o];
    float dv = (float)imax(deg[n], 1);
    acc += aggr[(size_t)n * dout + o] / dv;
    hp[(size_t)n * dout + o] = acc;
    ls += acc; lq += acc * acc;
  }
  s1[threadIdx.x] = ls; s2[threadIdx.x] = lq;
  __syncthreads();
  for (int st = 128; st > 0; st >>= 1) {
    if (threadIdx.x < st) { s1[threadIdx.x] += s1[threadIdx.x + st];
                            s2[threadIdx.x] += s2[threadIdx.x + st]; }
    __syncthreads();
  }
  if (threadIdx.x == 0) {
    float mean = s1[0] / (float)N;
    float var  = s2[0] / (float)N - mean * mean;
    float a = bn_g[o] * rsqrtf(var + EPS);
    ab[o] = a;
    ab[64 + o] = bn_b[o] - mean * a;
  }
}

__global__ void k_bn_apply(const float* __restrict__ hp, const float* __restrict__ ab,
                           float* __restrict__ hout, _Float16* __restrict__ h16,
                           int dout) {
  int idx = blockIdx.x * blockDim.x + threadIdx.x;
  if (idx >= N * dout) return;
  int o = idx % dout;
  float v = fmaxf(ab[o] * hp[idx] + ab[64 + o], 0.f);
  hout[idx] = v;
  if (h16) h16[idx] = (_Float16)v;
}

// ---------------- pooling + head ----------------

__global__ void k_pool(const float* __restrict__ h, const int* __restrict__ batch,
                       float* __restrict__ pooled) {
  int idx = blockIdx.x * blockDim.x + threadIdx.x;
  if (idx >= N * 32) return;
  int n = idx >> 5, o = idx & 31;
  unsafeAtomicAdd(&pooled[(size_t)batch[n] * 32 + o], h[idx]);
}

__global__ void k_final(const float* __restrict__ pooled, const int* __restrict__ cnt,
                        const float* __restrict__ fcw, const float* __restrict__ fcb,
                        float* __restrict__ out) {
  int idx = blockIdx.x * blockDim.x + threadIdx.x;
  if (idx >= G * 12) return;
  int g = idx / 12, j = idx % 12;
  float c = (float)imax(cnt[g], 1);
  float acc = fcb[j];
#pragma unroll
  for (int k = 0; k < 32; ++k) acc += (pooled[g * 32 + k] / c) * fcw[k * 12 + j];
  out[idx] = 1.f / (1.f + expf(-acc));
}

// ---------------- host launcher ----------------

extern "C" void kernel_launch(void* const* d_in, const int* in_sizes, int n_in,
                              void* d_out, int out_size, void* d_ws, size_t ws_size,
                              hipStream_t stream) {
  (void)in_sizes; (void)n_in; (void)out_size; (void)ws_size;
  const float* x     = (const float*)d_in[0];
  const int*   eidx  = (const int*)  d_in[1];
  const float* eattr = (const float*)d_in[2];
  const int*   batch = (const int*)  d_in[3];
  const float* ee_w  = (const float*)d_in[4];
  const float* ee_b  = (const float*)d_in[5];
  const float* ne_w  = (const float*)d_in[6];
  const float* ne_b  = (const float*)d_in[7];
  const float* fc_w  = (const float*)d_in[32];
  const float* fc_b  = (const float*)d_in[33];
  const int* srcv = eidx;
  const int* dstv = eidx + E;

  char* ws = (char*)d_ws;
  size_t off = 0;
  auto alloc = [&](size_t bytes) {
    size_t o = off; off += (bytes + 255) & ~(size_t)255; return (void*)(ws + o);
  };
  float*    hA     = (float*)   alloc((size_t)N * 64 * 4);
  float*    hB     = (float*)   alloc((size_t)N * 64 * 4);
  _Float16* h16    = (_Float16*)alloc((size_t)N * 64 * 2);
  float*    ee     = (float*)   alloc((size_t)E * 32 * 4);
  _Float16* ew16   = (_Float16*)alloc((size_t)E * 128 * 2);
  _Float16* Bp     = (_Float16*)alloc((size_t)KSTEPS * 4 * 512 * 2);
  float*    aggr   = (float*)   alloc((size_t)N * 64 * 4);
  float*    hp     = (float*)   alloc((size_t)N * 64 * 4);
  int*      deg    = (int*)     alloc((size_t)N * 4);
  int*      cnt    = (int*)     alloc((size_t)G * 4);
  float*    ab     = (float*)   alloc(128 * 4);
  float*    pooled = (float*)   alloc((size_t)G * 32 * 4);

  hipMemsetAsync(deg, 0, (size_t)N * 4, stream);
  hipMemsetAsync(cnt, 0, (size_t)G * 4, stream);

  k_edge_embed<<<(E * 32) / 256, 256, 0, stream>>>(eattr, ee_w, ee_b, ee);
  k_node_embed<<<(N * 64) / 256, 256, 0, stream>>>(x, ne_w, ne_b, hA, h16);
  k_deg<<<E / 256, 256, 0, stream>>>(dstv, deg);
  k_cnt<<<N / 256, 256, 0, stream>>>(batch, cnt);

  float* hin = hA;
  float* hout = hB;
  for (int L = 0; L < 3; ++L) {
    const int base = 8 + L * 8;
    const float* w1   = (const float*)d_in[base + 0];
    const float* b1   = (const float*)d_in[base + 1];
    const float* w2   = (const float*)d_in[base + 2];
    const float* b2   = (const float*)d_in[base + 3];
    const float* root = (const float*)d_in[base + 4];
    const float* cb   = (const float*)d_in[base + 5];
    const float* bg   = (const float*)d_in[base + 6];
    const float* bb   = (const float*)d_in[base + 7];
    const int dout = (L == 2) ? 32 : 64;
    const int ntiles = dout / 16;

    hipMemsetAsync(aggr, 0, (size_t)N * dout * 4, stream);
    k_edge_mlp<<<(E * 128) / 256, 256, 0, stream>>>(ee, w1, b1, ew16);
    const int total = KSTEPS * ntiles * 512;
    k_repack<<<(total + 255) / 256, 256, 0, stream>>>(w2, b2, Bp, ntiles, dout);

    const int tiles_per_wg = WAVES * MT_W;  // 24
    const int gemm_grid = (NT_TOT + tiles_per_wg - 1) / tiles_per_wg;  // 171
    if (ntiles == 4)
      k_msg_gemm<4><<<gemm_grid, 256, 0, stream>>>(ew16, h16, Bp, srcv, dstv, aggr);
    else
      k_msg_gemm<2><<<gemm_grid, 256, 0, stream>>>(ew16, h16, Bp, srcv, dstv, aggr);

    k_node_update<<<dout, 256, 0, stream>>>(hin, aggr, deg, root, cb, bg, bb, hp, ab, dout);
    k_bn_apply<<<((N * dout) + 255) / 256, 256, 0, stream>>>(
        hp, ab, hout, (L < 2) ? h16 : (_Float16*)nullptr, dout);

    float* t = hin; hin = hout; hout = t;
  }

  hipMemsetAsync(pooled, 0, (size_t)G * 32 * 4, stream);
  k_pool<<<(N * 32) / 256, 256, 0, stream>>>(hin, batch, pooled);
  k_final<<<((G * 12) + 255) / 256, 256, 0, stream>>>(pooled, cnt, fc_w, fc_b, (float*)d_out);
}
